// BAE_4475355922965
// MI455X (gfx1250) — matrix operand order repaired and checked by reading, not compiled, not run on hardware
//
#include <hip/hip_runtime.h>
#include <hip/hip_bf16.h>

typedef __attribute__((ext_vector_type(16))) _Float16 v16h;
typedef __attribute__((ext_vector_type(8)))  _Float16 v8h;
typedef __attribute__((ext_vector_type(4)))  _Float16 v4h;
typedef __attribute__((ext_vector_type(8)))  float    v8f;

#define BM 128
#define BN 128
#define KT 32

enum { EP_BIAS = 0, EP_RELU_BN = 1, EP_TANH = 2, EP_SPLIT = 3 };

union FragH { v16h v; v8h h[2]; };

// Generic f16-WMMA GEMM: C[M x N] = epilogue(A[M x K] @ W[K x N] + bias)
// A, W are f32 in global memory; converted to f16 while staging into LDS.
// Block tile 128x128, 8 waves (4 in M x 2 in N); each wave: 32x64 = 2x4 WMMA tiles.
// Software-pipelined: next K-tile is prefetched into registers during WMMA.
__global__ __launch_bounds__(256) void gemm_wmma_kernel(
    const float* __restrict__ A, const float* __restrict__ W,
    const float* __restrict__ bias,
    const float* __restrict__ bn_g, const float* __restrict__ bn_b,
    const float* __restrict__ bn_m, const float* __restrict__ bn_v,
    float* __restrict__ C, float* __restrict__ C2,
    int N, int K, int ldc, int mode)
{
    __shared__ __align__(16) _Float16 sA[BM * KT];   // row-major [row][k]
    __shared__ __align__(16) _Float16 sB[BN * KT];   // N-major  [col][k]

    const int t     = threadIdx.x;
    const int lane  = t & 31;
    const int wave  = t >> 5;
    const int waveM = wave >> 1;      // 0..3 -> 32 rows each
    const int waveN = wave & 1;       // 0..1 -> 64 cols each
    const int laneM = lane & 15;
    const int half  = lane >> 4;
    const int rowBase = blockIdx.y * BM;
    const int colBase = blockIdx.x * BN;

    // ---- precompute per-thread staging offsets (all 32-bit) ----
    // A: 128x32 elems = 1024 float4; 4 per thread.
    int aGOff[4], aLOff[4];
    #pragma unroll
    for (int i = 0; i < 4; ++i) {
        int idx4 = t + i * 256;
        int r  = idx4 >> 3;          // 0..127
        int kk = (idx4 & 7) << 2;    // 0,4,..,28
        aGOff[i] = r * K + kk;
        aLOff[i] = r * KT + kk;
    }
    // B: 32x128 elems; each thread handles 4 consecutive k for one column n.
    // Offsets clamped in-bounds; out-of-range columns masked to 0 at convert.
    int bGOff[4][4], bLOff[4];
    bool bOk[4];
    #pragma unroll
    for (int i = 0; i < 4; ++i) {
        int idx4 = t + i * 256;
        int n  = idx4 & 127;
        int kk = (idx4 >> 7) << 2;   // 0,4,..,28
        int gc = colBase + n;
        bOk[i] = (gc < N);
        int gcc = bOk[i] ? gc : (N - 1);
        #pragma unroll
        for (int d = 0; d < 4; ++d) bGOff[i][d] = (kk + d) * N + gcc;
        bLOff[i] = n * KT + kk;
    }
    const float* Abase = A + (size_t)rowBase * K;

    v8f acc[2][4];
    #pragma unroll
    for (int mi = 0; mi < 2; ++mi)
      #pragma unroll
      for (int ni = 0; ni < 4; ++ni)
        #pragma unroll
        for (int j = 0; j < 8; ++j) acc[mi][ni][j] = 0.0f;

    float4 aReg[4];
    float  bReg[4][4];
    auto prefetch = [&](int k0) {
        #pragma unroll
        for (int i = 0; i < 4; ++i)
            aReg[i] = *(const float4*)(Abase + (aGOff[i] + k0));
        int k0N = k0 * N;
        #pragma unroll
        for (int i = 0; i < 4; ++i)
            #pragma unroll
            for (int d = 0; d < 4; ++d)
                bReg[i][d] = W[bGOff[i][d] + k0N];
    };

    prefetch(0);

    for (int k0 = 0; k0 < K; k0 += KT) {
        // commit prefetched tile to LDS (f32 -> f16)
        #pragma unroll
        for (int i = 0; i < 4; ++i) {
            v4h h;
            h.x = (_Float16)aReg[i].x; h.y = (_Float16)aReg[i].y;
            h.z = (_Float16)aReg[i].z; h.w = (_Float16)aReg[i].w;
            *(v4h*)&sA[aLOff[i]] = h;
        }
        #pragma unroll
        for (int i = 0; i < 4; ++i) {
            v4h h;
            #pragma unroll
            for (int d = 0; d < 4; ++d)
                h[d] = (_Float16)(bOk[i] ? bReg[i][d] : 0.0f);
            *(v4h*)&sB[bLOff[i]] = h;
        }
        __syncthreads();

        // kick off next tile's global loads; they overlap with the WMMAs below
        int k1 = k0 + KT;
        if (k1 < K) prefetch(k1);

        FragH a[2], b[4];
        // A fragment: lane holds row (lane&15),
        // k = half*16 .. +15 contiguous (matches B)
        #pragma unroll
        for (int mi = 0; mi < 2; ++mi) {
            int r = waveM * 32 + mi * 16 + laneM;
            a[mi].h[0] = *(const v8h*)&sA[r * KT + half * 16];
            a[mi].h[1] = *(const v8h*)&sA[r * KT + half * 16 + 8];
        }
        // B fragment: lane holds col (lane&15), k = half*16 .. +15 contiguous
        #pragma unroll
        for (int ni = 0; ni < 4; ++ni) {
            int c = waveN * 64 + ni * 16 + laneM;
            b[ni].h[0] = *(const v8h*)&sB[c * KT + half * 16];
            b[ni].h[1] = *(const v8h*)&sB[c * KT + half * 16 + 8];
        }
        #pragma unroll
        for (int mi = 0; mi < 2; ++mi)
          #pragma unroll
          for (int ni = 0; ni < 4; ++ni)
            acc[mi][ni] = __builtin_amdgcn_wmma_f32_16x16x32_f16(
                false, a[mi].v, false, b[ni].v, (short)0, acc[mi][ni],
                false, false);
        __syncthreads();
    }

    // epilogue: C/D layout -> VGPR j holds row (half*8 + j), col = lane&15
    #pragma unroll
    for (int mi = 0; mi < 2; ++mi) {
      #pragma unroll
      for (int ni = 0; ni < 4; ++ni) {
        int col = colBase + waveN * 64 + ni * 16 + laneM;
        if (col >= N) continue;
        float bv = bias ? bias[col] : 0.0f;
        float g = 0.f, bb = 0.f, mm = 0.f, rv = 0.f;
        if (mode == EP_RELU_BN) {
            g  = bn_g[col]; bb = bn_b[col]; mm = bn_m[col];
            rv = rsqrtf(bn_v[col] + 1e-5f);
        }
        int rbase = rowBase + waveM * 32 + mi * 16 + half * 8;
        #pragma unroll
        for (int j = 0; j < 8; ++j) {
            int row = rbase + j;
            float v = acc[mi][ni][j] + bv;
            if (mode == EP_RELU_BN) { v = fmaxf(v, 0.0f); v = (v - mm) * rv * g + bb; }
            else if (mode == EP_TANH) v = tanhf(v);
            if (mode == EP_SPLIT) {
                if (col < 900) C [(size_t)row * 900 + col]        = v;
                else           C2[(size_t)row * 150 + (col - 900)] = v;
            } else {
                C[(size_t)row * ldc + col] = v;
            }
        }
      }
    }
}

// Per-batch-row: 4 softmaxes + in-place scaling of the OPT nodes/edges.
__global__ void attn_scale_kernel(float* __restrict__ out,
    const float* __restrict__ nw1, const float* __restrict__ ew1,
    const float* __restrict__ nw2, const float* __restrict__ ew2, int B)
{
    int b = blockIdx.x * blockDim.x + threadIdx.x;
    if (b >= B) return;
    float na[10], ea[55];
    {   // softmax over 10 (optical nodes), keep all 10
        const float* p = nw1 + (size_t)b * 16;
        float mx = -3.4e38f;
        for (int i = 0; i < 10; ++i) mx = fmaxf(mx, p[i]);
        float w[10], s = 0.f;
        for (int i = 0; i < 10; ++i) { w[i] = expf(p[i] - mx); s += w[i]; }
        for (int i = 0; i < 10; ++i) na[i] = w[i] / s;
    }
    {   // softmax over 12 (ot nodes), keep first 10
        const float* p = nw2 + (size_t)b * 16;
        float mx = -3.4e38f;
        for (int i = 0; i < 12; ++i) mx = fmaxf(mx, p[i]);
        float w[12], s = 0.f;
        for (int i = 0; i < 12; ++i) { w[i] = expf(p[i] - mx); s += w[i]; }
        for (int i = 0; i < 10; ++i) na[i] *= w[i] / s;
    }
    {   // softmax over 55 (optical edges), keep all 55
        const float* p = ew1 + (size_t)b * 64;
        float mx = -3.4e38f;
        for (int i = 0; i < 55; ++i) mx = fmaxf(mx, p[i]);
        float w[55], s = 0.f;
        for (int i = 0; i < 55; ++i) { w[i] = expf(p[i] - mx); s += w[i]; }
        for (int i = 0; i < 55; ++i) ea[i] = w[i] / s;
    }
    {   // softmax over 78 (ot edges), keep first 55
        const float* p = ew2 + (size_t)b * 80;
        float mx = -3.4e38f;
        for (int i = 0; i < 78; ++i) mx = fmaxf(mx, p[i]);
        float w[78], s = 0.f;
        for (int i = 0; i < 78; ++i) { w[i] = expf(p[i] - mx); s += w[i]; }
        for (int i = 0; i < 55; ++i) ea[i] *= w[i] / s;
    }
    // node[b, 20+n, :] *= na[n]
    float* node = out + (size_t)B * 900 + (size_t)b * 150;
    for (int n = 0; n < 10; ++n) {
        float s = na[n];
        for (int f = 0; f < 5; ++f) node[(20 + n) * 5 + f] *= s;
    }
    // edge[b, i, j] *= ea[e] for upper-triangular pairs over nodes 20..29
    float* edge = out + (size_t)b * 900;
    int e = 0;
    for (int i = 20; i < 30; ++i)
        for (int j = i; j < 30; ++j) { edge[i * 30 + j] *= ea[e]; ++e; }
}

extern "C" void kernel_launch(void* const* d_in, const int* in_sizes, int n_in,
                              void* d_out, int out_size, void* d_ws, size_t ws_size,
                              hipStream_t stream)
{
    const float* x       = (const float*)d_in[0];
    const float* optical = (const float*)d_in[1];
    const float* log_x   = (const float*)d_in[2];
    const float* enc_W   = (const float*)d_in[3];  const float* enc_b  = (const float*)d_in[4];
    const float* bn1_g   = (const float*)d_in[5];  const float* bn1_b  = (const float*)d_in[6];
    const float* bn1_m   = (const float*)d_in[7];  const float* bn1_v  = (const float*)d_in[8];
    const float* mu_W    = (const float*)d_in[9];  const float* mu_b   = (const float*)d_in[10];
    const float* lv_W    = (const float*)d_in[11]; const float* lv_b   = (const float*)d_in[12];
    const float* dec_W1  = (const float*)d_in[13]; const float* dec_b1 = (const float*)d_in[14];
    const float* bn2_g   = (const float*)d_in[15]; const float* bn2_b  = (const float*)d_in[16];
    const float* bn2_m   = (const float*)d_in[17]; const float* bn2_v  = (const float*)d_in[18];
    const float* dec_W2  = (const float*)d_in[19]; const float* dec_b2 = (const float*)d_in[20];
    const float* on_W1   = (const float*)d_in[21]; const float* on_b1  = (const float*)d_in[22];
    const float* on_W2   = (const float*)d_in[23]; const float* on_b2  = (const float*)d_in[24];
    const float* oe_W1   = (const float*)d_in[25]; const float* oe_b1  = (const float*)d_in[26];
    const float* oe_W2   = (const float*)d_in[27]; const float* oe_b2  = (const float*)d_in[28];
    const float* ln_W1   = (const float*)d_in[29]; const float* ln_b1  = (const float*)d_in[30];
    const float* ln_W2   = (const float*)d_in[31]; const float* ln_b2  = (const float*)d_in[32];
    const float* le_W1   = (const float*)d_in[33]; const float* le_b1  = (const float*)d_in[34];
    const float* le_W2   = (const float*)d_in[35]; const float* le_b2  = (const float*)d_in[36];

    const int Bn = in_sizes[0] / 1024;   // 16384
    const int D = 1024, H = 2048, L = 256, OUTD = 1050;

    float* outp   = (float*)d_out;
    float* edge   = outp;                              // [B,30,30]
    float* node   = outp + (size_t)Bn * 900;           // [B,30,5]
    float* mu     = outp + (size_t)Bn * 1050;          // [B,256]
    float* logvar = outp + (size_t)Bn * 1306;          // [B,256]

    float* ws   = (float*)d_ws;
    float* hBuf = ws;                                  // B x 2048 (h1, then h2)
    float* tBuf = ws + (size_t)Bn * H;                 // B x 512 (tanh hidden, reused)
    float* nw1  = tBuf + (size_t)Bn * 512;             // B x 16 (10 used)
    float* ew1  = nw1 + (size_t)Bn * 16;               // B x 64 (55 used)
    float* nw2  = ew1 + (size_t)Bn * 64;               // B x 16 (12 used)
    float* ew2  = nw2 + (size_t)Bn * 16;               // B x 80 (78 used)

    dim3 blk(256);
    auto grd = [&](int Ncols) { return dim3((Ncols + BN - 1) / BN, Bn / BM); };

    // h1 = BN(relu(x @ enc_W + enc_b))
    gemm_wmma_kernel<<<grd(H), blk, 0, stream>>>(x, enc_W, enc_b, bn1_g, bn1_b, bn1_m, bn1_v,
                                                 hBuf, nullptr, H, D, H, EP_RELU_BN);
    // mu, logvar
    gemm_wmma_kernel<<<grd(L), blk, 0, stream>>>(hBuf, mu_W, mu_b, nullptr, nullptr, nullptr, nullptr,
                                                 mu, nullptr, L, H, L, EP_BIAS);
    gemm_wmma_kernel<<<grd(L), blk, 0, stream>>>(hBuf, lv_W, lv_b, nullptr, nullptr, nullptr, nullptr,
                                                 logvar, nullptr, L, H, L, EP_BIAS);
    // h2 = BN(relu(mu @ dec_W1 + dec_b1))  (z == mu in eval mode)
    gemm_wmma_kernel<<<grd(H), blk, 0, stream>>>(mu, dec_W1, dec_b1, bn2_g, bn2_b, bn2_m, bn2_v,
                                                 hBuf, nullptr, H, L, H, EP_RELU_BN);
    // out = h2 @ dec_W2 + dec_b2 -> split into edge/node sections
    gemm_wmma_kernel<<<grd(OUTD), blk, 0, stream>>>(hBuf, dec_W2, dec_b2, nullptr, nullptr, nullptr, nullptr,
                                                    edge, node, OUTD, H, 0, EP_SPLIT);
    // four attention MLPs (tanh hidden 512, then small output layer)
    gemm_wmma_kernel<<<grd(512), blk, 0, stream>>>(optical, on_W1, on_b1, nullptr, nullptr, nullptr, nullptr,
                                                   tBuf, nullptr, 512, D, 512, EP_TANH);
    gemm_wmma_kernel<<<grd(10),  blk, 0, stream>>>(tBuf, on_W2, on_b2, nullptr, nullptr, nullptr, nullptr,
                                                   nw1, nullptr, 10, 512, 16, EP_BIAS);
    gemm_wmma_kernel<<<grd(512), blk, 0, stream>>>(optical, oe_W1, oe_b1, nullptr, nullptr, nullptr, nullptr,
                                                   tBuf, nullptr, 512, D, 512, EP_TANH);
    gemm_wmma_kernel<<<grd(55),  blk, 0, stream>>>(tBuf, oe_W2, oe_b2, nullptr, nullptr, nullptr, nullptr,
                                                   ew1, nullptr, 55, 512, 64, EP_BIAS);
    gemm_wmma_kernel<<<grd(512), blk, 0, stream>>>(log_x, ln_W1, ln_b1, nullptr, nullptr, nullptr, nullptr,
                                                   tBuf, nullptr, 512, D, 512, EP_TANH);
    gemm_wmma_kernel<<<grd(12),  blk, 0, stream>>>(tBuf, ln_W2, ln_b2, nullptr, nullptr, nullptr, nullptr,
                                                   nw2, nullptr, 12, 512, 16, EP_BIAS);
    gemm_wmma_kernel<<<grd(512), blk, 0, stream>>>(log_x, le_W1, le_b1, nullptr, nullptr, nullptr, nullptr,
                                                   tBuf, nullptr, 512, D, 512, EP_TANH);
    gemm_wmma_kernel<<<grd(78),  blk, 0, stream>>>(tBuf, le_W2, le_b2, nullptr, nullptr, nullptr, nullptr,
                                                   ew2, nullptr, 78, 512, 80, EP_BIAS);

    attn_scale_kernel<<<dim3((Bn + 255) / 256), blk, 0, stream>>>(outp, nw1, ew1, nw2, ew2, Bn);
}
